// HolographicMemoryNetwork_37649683316882
// MI455X (gfx1250) — matrix operand order repaired and checked by reading, not compiled, not run on hardware
//
#include <hip/hip_runtime.h>
#include <hip/hip_fp16.h>

typedef __attribute__((ext_vector_type(16))) _Float16 v16h;
typedef __attribute__((ext_vector_type(8)))  float    v8f;

#define DD    256
#define NH    8
#define HDIM  32
#define TOPK  10
#define CAPN  500000
#define BB    256
#define NB1   245          // blocks for sims kernel
#define CHUNK 2048         // hologram rows per block (245*2048 >= 500000)
#define NT1   (CHUNK/16)   // 128 N-tiles per block
#define LH    264          // LDS hologram row stride (f16 elems), padded
#define LS    20           // LDS sims row stride (f32), 16B-aligned + conflict-light

__device__ __forceinline__ bool kbetter(float av, int ai, float bv, int bi) {
    return (av > bv) || (av == bv && ai < bi);
}

// ---------------------------------------------------------------------------
// Kernel 1: sims = content @ holo^T * strength, fused per-block top-10.
// Block = 256 threads = 8 waves. Wave w owns content rows [32w, 32w+32).
// A fragments (content, f16) live in registers for the whole block.
// Hologram tiles are double-buffered in LDS; the global loads for tile t+1
// are issued before the WMMA burst of tile t (software pipeline), so HBM
// latency overlaps compute. One __syncthreads per tile.
// ---------------------------------------------------------------------------
__global__ __launch_bounds__(256) void k_sims_topk(
    const float* __restrict__ content, const float* __restrict__ holo,
    const float* __restrict__ strength,
    float* __restrict__ pvals, int* __restrict__ pidx)
{
    __shared__ _Float16 hT[2][16 * LH];
    __shared__ float    simsS[8 * 32 * LS];

    const int tid  = threadIdx.x;
    const int lane = tid & 31;
    const int wave = tid >> 5;
    const int l15  = lane & 15;
    const int hi16 = lane >> 4;

    // Preload A fragments: 2 M-tiles x 8 K-steps, f32 -> f16, registers.
    v16h afr[2][8];
    {
        const int kb = hi16 << 3;
        #pragma unroll
        for (int mt = 0; mt < 2; ++mt) {
            const float* crow = content + (wave * 32 + mt * 16 + l15) * DD;
            #pragma unroll
            for (int ks = 0; ks < 8; ++ks) {
                const int k0 = ks * 32 + kb;
                #pragma unroll
                for (int j = 0; j < 8; ++j) {
                    afr[mt][ks][j]     = (_Float16)crow[k0 + j];
                    afr[mt][ks][8 + j] = (_Float16)crow[k0 + 16 + j];
                }
            }
        }
    }

    float tv[TOPK]; int ti[TOPK];
    #pragma unroll
    for (int j = 0; j < TOPK; ++j) { tv[j] = -__builtin_inff(); ti[j] = 0x7fffffff; }

    const long blockBase = (long)blockIdx.x * CHUNK;
    const int  srow = tid >> 4;         // staging row 0..15
    const int  scol = (tid & 15) * 16;  // staging 16-col group

    float4 st[4];

    // Prologue: stage tile 0 into buffer 0.
    {
        const long hr = blockBase + srow;
        if (hr < CAPN) {
            const float* src = holo + hr * (long)DD + scol;
            #pragma unroll
            for (int j = 0; j < 4; ++j) st[j] = *(const float4*)(src + 4 * j);
        } else {
            #pragma unroll
            for (int j = 0; j < 4; ++j) st[j] = make_float4(0.f, 0.f, 0.f, 0.f);
        }
        _Float16* dst = &hT[0][srow * LH + scol];
        #pragma unroll
        for (int j = 0; j < 4; ++j) {
            dst[4 * j + 0] = (_Float16)st[j].x; dst[4 * j + 1] = (_Float16)st[j].y;
            dst[4 * j + 2] = (_Float16)st[j].z; dst[4 * j + 3] = (_Float16)st[j].w;
        }
    }

    for (int t = 0; t < NT1; ++t) {
        __syncthreads();   // buf[t&1] staged by all; previous readers of buf[(t+1)&1] done

        const long n0 = blockBase + (long)t * 16;
        const bool have_next = (t + 1 < NT1);

        // Issue global loads for tile t+1 now; waitcnt sinks below the WMMAs.
        if (have_next) {
            const long hr = n0 + 16 + srow;
            if (hr < CAPN) {
                const float* src = holo + hr * (long)DD + scol;
                #pragma unroll
                for (int j = 0; j < 4; ++j) st[j] = *(const float4*)(src + 4 * j);
            } else {
                #pragma unroll
                for (int j = 0; j < 4; ++j) st[j] = make_float4(0.f, 0.f, 0.f, 0.f);
            }
            if (t + 2 < NT1) {  // prefetch tile t+2 -> global_prefetch_b8
                const long pr = n0 + 32 + srow;
                if (pr < CAPN) __builtin_prefetch(holo + pr * (long)DD + scol, 0, 1);
            }
        }

        const long hn    = n0 + l15;
        const bool valid = hn < CAPN;
        const float sc   = valid ? strength[hn] : 0.f;

        // WMMA burst on tile t.
        const _Float16* hbuf = hT[t & 1];
        float* sb = &simsS[wave * 32 * LS];
        #pragma unroll
        for (int mt = 0; mt < 2; ++mt) {
            v8f acc = {};
            #pragma unroll
            for (int ks = 0; ks < 8; ++ks) {
                const _Float16* bp = &hbuf[l15 * LH + ks * 32 + (hi16 << 3)];
                v16h bfr;
                #pragma unroll
                for (int j = 0; j < 16; ++j) bfr[j] = bp[j < 8 ? j : j + 8];
                acc = __builtin_amdgcn_wmma_f32_16x16x32_f16(
                          false, afr[mt][ks], false, bfr, (short)0, acc, false, false);
            }
            const int rb = mt * 16 + hi16 * 8;
            #pragma unroll
            for (int i = 0; i < 8; ++i)
                sb[(rb + i) * LS + l15] = valid ? acc[i] * sc : -__builtin_inff();
        }

        // Scatter -> scan stays within this wave's simsS region; same-wave LDS
        // ops are executed in order, no block barrier needed.
        __builtin_amdgcn_wave_barrier();
        asm volatile("" ::: "memory");

        const float* row = &simsS[(wave * 32 + lane) * LS];
        #pragma unroll
        for (int j = 0; j < 16; ++j) {
            const float v = row[j];
            if (v > tv[0]) {
                tv[0] = v; ti[0] = (int)(n0 + j);
                #pragma unroll
                for (int q = 0; q < TOPK - 1; ++q) {
                    if (tv[q] > tv[q + 1]) {
                        float fv = tv[q]; tv[q] = tv[q + 1]; tv[q + 1] = fv;
                        int   fi = ti[q]; ti[q] = ti[q + 1]; ti[q + 1] = fi;
                    }
                }
            }
        }

        // Convert + store staged tile t+1 into the other LDS buffer.
        if (have_next) {
            _Float16* dst = &hT[(t + 1) & 1][srow * LH + scol];
            #pragma unroll
            for (int j = 0; j < 4; ++j) {
                dst[4 * j + 0] = (_Float16)st[j].x; dst[4 * j + 1] = (_Float16)st[j].y;
                dst[4 * j + 2] = (_Float16)st[j].z; dst[4 * j + 3] = (_Float16)st[j].w;
            }
        }
    }

    const int  r    = wave * 32 + lane;
    const long base = ((long)blockIdx.x * BB + r) * TOPK;
    #pragma unroll
    for (int j = 0; j < TOPK; ++j) {
        pvals[base + j] = tv[TOPK - 1 - j];
        pidx [base + j] = ti[TOPK - 1 - j];
    }
}

// ---------------------------------------------------------------------------
// Kernel 2: merge NB1 partial top-10 lists per row -> final top-10 indices.
// One block per content row, 64 threads, jax tie-break (val desc, idx asc).
// ---------------------------------------------------------------------------
__global__ __launch_bounds__(64) void k_topk_reduce(
    const float* __restrict__ pv, const int* __restrict__ pi,
    int* __restrict__ outIdx)
{
    __shared__ float lv[64 * TOPK];
    __shared__ int   li[64 * TOPK];
    const int b = blockIdx.x, t = threadIdx.x;

    float tv[TOPK]; int ti[TOPK];
    #pragma unroll
    for (int j = 0; j < TOPK; ++j) { tv[j] = -__builtin_inff(); ti[j] = 0x7fffffff; }

    const int total = NB1 * TOPK;
    for (int c = t; c < total; c += 64) {
        const int  cb = c / TOPK, cj = c - cb * TOPK;
        const long o  = ((long)cb * BB + b) * TOPK + cj;
        const float v = pv[o]; const int id = pi[o];
        if (kbetter(v, id, tv[0], ti[0])) {
            tv[0] = v; ti[0] = id;
            #pragma unroll
            for (int q = 0; q < TOPK - 1; ++q) {
                if (kbetter(tv[q], ti[q], tv[q + 1], ti[q + 1])) {
                    float fv = tv[q]; tv[q] = tv[q + 1]; tv[q + 1] = fv;
                    int   fi = ti[q]; ti[q] = ti[q + 1]; ti[q + 1] = fi;
                }
            }
        }
    }
    #pragma unroll
    for (int j = 0; j < TOPK; ++j) { lv[t * TOPK + j] = tv[j]; li[t * TOPK + j] = ti[j]; }
    __syncthreads();

    if (t == 0) {
        float fv[TOPK]; int fi[TOPK];
        #pragma unroll
        for (int j = 0; j < TOPK; ++j) { fv[j] = -__builtin_inff(); fi[j] = 0x7fffffff; }
        for (int c = 0; c < 64 * TOPK; ++c) {
            const float v = lv[c]; const int id = li[c];
            if (kbetter(v, id, fv[0], fi[0])) {
                fv[0] = v; fi[0] = id;
                #pragma unroll
                for (int q = 0; q < TOPK - 1; ++q) {
                    if (kbetter(fv[q], fi[q], fv[q + 1], fi[q + 1])) {
                        float a = fv[q]; fv[q] = fv[q + 1]; fv[q + 1] = a;
                        int   x = fi[q]; fi[q] = fi[q + 1]; fi[q + 1] = x;
                    }
                }
            }
        }
        #pragma unroll
        for (int j = 0; j < TOPK; ++j) outIdx[b * TOPK + j] = fi[TOPK - 1 - j];
    }
}

// ---------------------------------------------------------------------------
// Helpers: f32->f16 convert, and gather hologram rows -> f16 "mem" matrix.
// ---------------------------------------------------------------------------
__global__ void k_f32_to_f16(const float* __restrict__ s, _Float16* __restrict__ d, int n) {
    const int i = blockIdx.x * 256 + threadIdx.x;
    if (i < n) d[i] = (_Float16)s[i];
}

__global__ void k_gather(const float* __restrict__ holo, const int* __restrict__ idx,
                         _Float16* __restrict__ memf) {
    const int  r = blockIdx.x, c = threadIdx.x;  // grid 2560 x 256
    const long h = idx[r];
    memf[(long)r * DD + c] = (_Float16)holo[h * (long)DD + c];
}

// ---------------------------------------------------------------------------
// Generic WMMA GEMM: out[M][256] = A_f16[M][256] @ W_f16[256][256]^T + bias.
// One 16x16 tile per wave, 8 waves per block; grid = (M/16)*16/8 blocks.
// ---------------------------------------------------------------------------
__global__ __launch_bounds__(256) void k_proj(
    const _Float16* __restrict__ A, const _Float16* __restrict__ W,
    const float* __restrict__ bias, float* __restrict__ out)
{
    const int lane = threadIdx.x & 31;
    const int wave = threadIdx.x >> 5;
    const int l15  = lane & 15;
    const int hi16 = lane >> 4;
    const int tile = blockIdx.x * 8 + wave;
    const int mt   = tile >> 4;   // 16 N-tiles per M-tile (256/16)
    const int nt   = tile & 15;

    const _Float16* arow = A + ((long)mt * 16 + l15) * DD + (hi16 << 3);
    const _Float16* wrow = W + ((long)nt * 16 + l15) * DD + (hi16 << 3);

    v8f acc = {};
    #pragma unroll
    for (int ks = 0; ks < 8; ++ks) {
        v16h a, bfr;
        #pragma unroll
        for (int j = 0; j < 8; ++j) {
            a[j]     = arow[ks * 32 + j];
            a[8 + j] = arow[ks * 32 + 16 + j];
        }
        #pragma unroll
        for (int j = 0; j < 16; ++j) bfr[j] = wrow[ks * 32 + (j < 8 ? j : j + 8)];
        acc = __builtin_amdgcn_wmma_f32_16x16x32_f16(
                  false, a, false, bfr, (short)0, acc, false, false);
    }
    const float bv = bias[nt * 16 + l15];
    const int   rb = hi16 * 8;
    #pragma unroll
    for (int i = 0; i < 8; ++i)
        out[((long)mt * 16 + rb + i) * DD + nt * 16 + l15] = acc[i] + bv;
}

// ---------------------------------------------------------------------------
// Attention: one block per b, one wave per head. K=10, HD=32 (= wave width).
// Writes ctx (f16) and attn_weights (mean over heads) directly to d_out tail.
// ---------------------------------------------------------------------------
__global__ __launch_bounds__(256) void k_attn(
    const float* __restrict__ qB, const float* __restrict__ kB,
    const float* __restrict__ vB, _Float16* __restrict__ ctx,
    float* __restrict__ attnOut)
{
    __shared__ float aw[NH][TOPK];
    const int b = blockIdx.x;
    const int h = threadIdx.x >> 5;
    const int d = threadIdx.x & 31;

    const float qv = qB[(long)b * DD + h * HDIM + d];
    float sc[TOPK];
    #pragma unroll
    for (int kk = 0; kk < TOPK; ++kk) {
        float p = qv * kB[(long)(b * TOPK + kk) * DD + h * HDIM + d];
        #pragma unroll
        for (int m = 16; m >= 1; m >>= 1) p += __shfl_xor(p, m, 32);
        sc[kk] = p * 0.17677669529663687f;   // 1/sqrt(32)
    }
    float mx = sc[0];
    #pragma unroll
    for (int kk = 1; kk < TOPK; ++kk) mx = fmaxf(mx, sc[kk]);
    float sum = 0.f;
    #pragma unroll
    for (int kk = 0; kk < TOPK; ++kk) { sc[kk] = __expf(sc[kk] - mx); sum += sc[kk]; }
    const float inv = 1.f / sum;

    float cv = 0.f;
    #pragma unroll
    for (int kk = 0; kk < TOPK; ++kk) {
        const float a = sc[kk] * inv;
        cv += a * vB[(long)(b * TOPK + kk) * DD + h * HDIM + d];
        if (d == 0) aw[h][kk] = a;
    }
    ctx[(long)b * DD + h * HDIM + d] = (_Float16)cv;
    __syncthreads();

    if (threadIdx.x < TOPK) {
        float s2 = 0.f;
        #pragma unroll
        for (int hh = 0; hh < NH; ++hh) s2 += aw[hh][threadIdx.x];
        attnOut[(long)b * TOPK + threadIdx.x] = s2 * (1.f / NH);
    }
}

// ---------------------------------------------------------------------------
extern "C" void kernel_launch(void* const* d_in, const int* in_sizes, int n_in,
                              void* d_out, int out_size, void* d_ws, size_t ws_size,
                              hipStream_t stream) {
    (void)in_sizes; (void)n_in; (void)out_size; (void)ws_size;
    const float* content   = (const float*)d_in[0];
    const float* hologram  = (const float*)d_in[1];
    const float* strengths = (const float*)d_in[2];
    const float* in_w      = (const float*)d_in[3];
    const float* in_b      = (const float*)d_in[4];
    const float* out_w     = (const float*)d_in[5];
    const float* out_b     = (const float*)d_in[6];

    size_t off = 0;
    char* ws = (char*)d_ws;
    auto carve = [&](size_t bytes) -> void* {
        void* p = ws + off; off += (bytes + 255) & ~(size_t)255; return p;
    };
    float*     pvals = (float*)    carve((size_t)NB1 * BB * TOPK * 4);
    int*       pidx  = (int*)      carve((size_t)NB1 * BB * TOPK * 4);
    int*       fidx  = (int*)      carve((size_t)BB * TOPK * 4);
    _Float16*  cF    = (_Float16*) carve((size_t)BB * DD * 2);
    _Float16*  wF    = (_Float16*) carve((size_t)3 * DD * DD * 2);
    _Float16*  oF    = (_Float16*) carve((size_t)DD * DD * 2);
    _Float16*  mF    = (_Float16*) carve((size_t)BB * TOPK * DD * 2);
    float*     qb    = (float*)    carve((size_t)BB * DD * 4);
    float*     kb    = (float*)    carve((size_t)BB * TOPK * DD * 4);
    float*     vb    = (float*)    carve((size_t)BB * TOPK * DD * 4);
    _Float16*  ctxF  = (_Float16*) carve((size_t)BB * DD * 2);

    // 1) dominant fused GEMM + per-block top-k
    k_sims_topk<<<NB1, 256, 0, stream>>>(content, hologram, strengths, pvals, pidx);
    // 2) final top-k indices
    k_topk_reduce<<<BB, 64, 0, stream>>>(pvals, pidx, fidx);
    // 3) f16 conversions + gather
    k_f32_to_f16<<<(BB * DD + 255) / 256, 256, 0, stream>>>(content, cF, BB * DD);
    k_f32_to_f16<<<(3 * DD * DD + 255) / 256, 256, 0, stream>>>(in_w, wF, 3 * DD * DD);
    k_f32_to_f16<<<(DD * DD + 255) / 256, 256, 0, stream>>>(out_w, oF, DD * DD);
    k_gather<<<BB * TOPK, 256, 0, stream>>>(hologram, fidx, mF);
    // 4) projections (WMMA)
    k_proj<<<32, 256, 0, stream>>>(cF, wF, in_b, qb);                         // q: M=256
    k_proj<<<320, 256, 0, stream>>>(mF, wF + DD * DD, in_b + DD, kb);         // k: M=2560
    k_proj<<<320, 256, 0, stream>>>(mF, wF + 2 * DD * DD, in_b + 2 * DD, vb); // v: M=2560
    // 5) attention -> ctx(f16) + attn_weights (d_out tail)
    k_attn<<<BB, 256, 0, stream>>>(qb, kb, vb, ctxF, (float*)d_out + BB * DD);
    // 6) out projection -> recalled (d_out head)
    k_proj<<<32, 256, 0, stream>>>(ctxF, oF, out_b, (float*)d_out);
}
